// _ALiBiMHA_9552007266472
// MI455X (gfx1250) — matrix operand order repaired and checked by reading, not compiled, not run on hardware
//
#include <hip/hip_runtime.h>
#include <hip/hip_bf16.h>

// ---------------------------------------------------------------------------
// ALiBi windowed MHA for MI455X (gfx1250, wave32, WMMA).
// B=2, T=2048, D=1024, H=16, dk=64, window=256 (band = +/-128).
//
// Pipeline:
//  0) memsetAsync attn region of d_out to 0 (attention is banded).
//  1) convert x, Wq, Wk, Wv, Wo  f32 -> bf16 (workspace)
//  2) WMMA GEMM projections: Q,(B,H,T,dk)  K,(B,H,T,dk)  V^T,(B,H,dk,T)
//  3) attention kernel: per (b,h,16-query tile) one wave:
//        S = Q K^T via v_wmma_f32_16x16x32_bf16 (x2 for dk=64)
//        + alibi + window mask, softmax via 16-lane butterflies + LDS staging
//        write banded attn probs (f32) to d_out, then O = P V via WMMA
//  4) WMMA GEMM: out = O @ Wo^T  (f32 result straight into d_out)
// ---------------------------------------------------------------------------

typedef __bf16 bf16_t;
typedef __attribute__((ext_vector_type(16))) bf16_t v16bf;
typedef __attribute__((ext_vector_type(8)))  bf16_t v8bf;
typedef __attribute__((ext_vector_type(8)))  float  v8f;

union AFrag { v16bf v; v8bf h[2]; };

__device__ __forceinline__ v8f wmma_bf16(v16bf a, v16bf b, v8f c) {
  // (neg_a, A, neg_b, B, c_mod, C, reuse_a, reuse_b)
  return __builtin_amdgcn_wmma_f32_16x16x32_bf16(false, a, false, b, (short)0, c,
                                                 false, false);
}

static constexpr int Bn  = 2;
static constexpr int Tn  = 2048;
static constexpr int Dn  = 1024;
static constexpr int Hn  = 16;
static constexpr int DKn = 64;

// --------------------------- f32 -> bf16 convert ---------------------------
__global__ __launch_bounds__(256) void cvt_f32_bf16(const float* __restrict__ src,
                                                    bf16_t* __restrict__ dst,
                                                    int n) {
  int i = blockIdx.x * blockDim.x + threadIdx.x;
  int stride = gridDim.x * blockDim.x;
  for (; i < n; i += stride) dst[i] = (bf16_t)src[i];
}

// ------------------------------- WMMA GEMM ---------------------------------
// C[m][o] = sum_k A[m][k] * W[o][k]     (A: MxK row-major, W: NxK row-major)
// Each wave computes a 16x64 tile (4 chained N-subtiles share the A fragment).
// mode 0: scatter bf16 to (B,H,T,dk)     (Q / K)
// mode 1: scatter bf16 to (B,H,dk,T)     (V transposed)
// mode 2: f32 row-major MxN              (final output)
__global__ __launch_bounds__(128) void proj_gemm(const bf16_t* __restrict__ A,
                                                 const bf16_t* __restrict__ W,
                                                 bf16_t* __restrict__ out_bf,
                                                 float* __restrict__ out_f,
                                                 int M, int N, int K, int mode) {
  const int ln  = threadIdx.x;          // lane 0..31
  const int hf  = ln >> 4;              // half-wave
  const int l15 = ln & 15;
  const int mtile = blockIdx.x;
  const int n64   = blockIdx.y * 4 + threadIdx.y;

  const int row_a = mtile * 16 + l15;

  v8f acc[4];
  for (int i = 0; i < 4; ++i)
    for (int j = 0; j < 8; ++j) acc[i][j] = 0.0f;

  const int nkb = K >> 5;
  for (int kb = 0; kb < nkb; ++kb) {
    // A fragment: one contiguous 16-elt run per lane (k = 16*hf + s).
    const bf16_t* ap = A + (size_t)row_a * K + kb * 32 + hf * 16;
    AFrag af;
    af.h[0] = *(const v8bf*)(ap);
    af.h[1] = *(const v8bf*)(ap + 8);
    __builtin_prefetch(ap + 32, 0, 0);  // global_prefetch_b8 for next K-block

#pragma unroll
    for (int nt = 0; nt < 4; ++nt) {
      const int col = n64 * 64 + nt * 16 + l15;
      // B fragment: one contiguous 16-elt run per lane.
      const bf16_t* bp = W + (size_t)col * K + kb * 32 + hf * 16;
      v16bf bv = *(const v16bf*)bp;
      acc[nt] = wmma_bf16(af.v, bv, acc[nt]);
    }
  }

  // write back (C/D layout: VGPR r -> row r + 8*hf, col = lane&15)
#pragma unroll
  for (int nt = 0; nt < 4; ++nt) {
#pragma unroll
    for (int r = 0; r < 8; ++r) {
      const int m = mtile * 16 + r + 8 * hf;
      const int o = n64 * 64 + nt * 16 + l15;
      const float v = acc[nt][r];
      if (mode == 2) {
        out_f[(size_t)m * N + o] = v;
      } else {
        const int b = m / Tn, t = m % Tn;
        const int h = o >> 6, d = o & 63;
        size_t idx;
        if (mode == 0)  // (B,H,T,dk)
          idx = (((size_t)b * Hn + h) * Tn + t) * DKn + d;
        else            // (B,H,dk,T)
          idx = (((size_t)b * Hn + h) * DKn + d) * Tn + t;
        out_bf[idx] = (bf16_t)v;
      }
    }
  }
}

// ------------------------------- Attention ---------------------------------
static constexpr int SMAXT = 18;             // max 16-key tiles in band
static constexpr int SCOLS = SMAXT * 16;     // 288

__global__ __launch_bounds__(64) void attn_kernel(const bf16_t* __restrict__ Qb,
                                                  const bf16_t* __restrict__ Kb,
                                                  const bf16_t* __restrict__ Vt,
                                                  bf16_t* __restrict__ Obuf,
                                                  float* __restrict__ attn_out,
                                                  const int* __restrict__ window_p) {
  __shared__ float Ssc[2][16][SCOLS];        // 36,864 B per workgroup

  const int ln  = threadIdx.x;
  const int wv  = threadIdx.y;               // which wave / which q-subtile
  const int hf  = ln >> 4;
  const int l15 = ln & 15;

  const int q0 = (blockIdx.x * 2 + wv) * 16;
  const int h  = blockIdx.y;
  const int b  = blockIdx.z;
  const int bh = b * Hn + h;

  const int   W2    = window_p[0] >> 1;
  const float slope = exp2f(-0.5f * (float)(h + 1));  // H=16 alibi slopes
  const float scale = rsqrtf((float)DKn);

  float (*S)[SCOLS] = Ssc[wv];

  // zero this wave's LDS slab (padding columns must be 0 for the PV matmuls)
  for (int i = ln; i < 16 * SCOLS; i += 32) ((float*)S)[i] = 0.0f;
  __syncthreads();

  // load Q fragment (16 rows x 64 K), 2 K-blocks of 32
  const int qrow = q0 + l15;
  const bf16_t* qbase = Qb + ((size_t)bh * Tn + qrow) * DKn;
  AFrag qf[2];
#pragma unroll
  for (int kb = 0; kb < 2; ++kb) {
    const bf16_t* qp = qbase + kb * 32 + hf * 16;
    qf[kb].h[0] = *(const v8bf*)(qp);
    qf[kb].h[1] = *(const v8bf*)(qp + 8);
  }

  // banded key range, 16-aligned
  int s_lo = q0 - W2;        if (s_lo < 0) s_lo = 0;
  int s_hi = q0 + 15 + W2;   if (s_hi > Tn - 1) s_hi = Tn - 1;
  const int st0 = s_lo >> 4;
  int ntiles = (s_hi >> 4) - st0 + 1;
  if (ntiles > SMAXT) ntiles = SMAXT;
  const int s_base = st0 << 4;

  float rmax[8];
#pragma unroll
  for (int r = 0; r < 8; ++r) rmax[r] = -__builtin_inff();

  // ---- pass 1: scores via WMMA, alibi + mask, stage in LDS, row max ----
  for (int j = 0; j < ntiles; ++j) {
    const int s0 = s_base + j * 16;
    const bf16_t* kbase = Kb + ((size_t)bh * Tn + (s0 + l15)) * DKn;
    v8f sc;
    for (int r = 0; r < 8; ++r) sc[r] = 0.0f;
#pragma unroll
    for (int kb = 0; kb < 2; ++kb) {
      v16bf bv = *(const v16bf*)(kbase + kb * 32 + hf * 16);
      sc = wmma_bf16(qf[kb].v, bv, sc);
    }
#pragma unroll
    for (int r = 0; r < 8; ++r) {
      const int q = q0 + r + 8 * hf;
      const int s = s0 + l15;
      const int dist = (q > s) ? (q - s) : (s - q);
      float v = sc[r] * scale - slope * (float)dist;
      if (dist > W2) v = -__builtin_inff();
      S[r + 8 * hf][j * 16 + l15] = v;
      rmax[r] = fmaxf(rmax[r], v);
    }
  }
  // row max across the 16 lanes holding one row
#pragma unroll
  for (int r = 0; r < 8; ++r) {
    for (int m = 1; m < 16; m <<= 1)
      rmax[r] = fmaxf(rmax[r], __shfl_xor(rmax[r], m, 16));
  }

  // ---- pass 2: exp + row sum ----
  float psum[8];
#pragma unroll
  for (int r = 0; r < 8; ++r) psum[r] = 0.0f;
  for (int j = 0; j < ntiles; ++j) {
#pragma unroll
    for (int r = 0; r < 8; ++r) {
      const int row = r + 8 * hf, col = j * 16 + l15;
      const float e = __expf(S[row][col] - rmax[r]);
      S[row][col] = e;
      psum[r] += e;
    }
  }
#pragma unroll
  for (int r = 0; r < 8; ++r) {
    for (int m = 1; m < 16; m <<= 1) psum[r] += __shfl_xor(psum[r], m, 16);
    psum[r] = 1.0f / psum[r];
  }

  // ---- pass 3: normalize in LDS + write banded attn probs (f32) ----
  for (int j = 0; j < ntiles; ++j) {
#pragma unroll
    for (int r = 0; r < 8; ++r) {
      const int row = r + 8 * hf, col = j * 16 + l15;
      const float p = S[row][col] * psum[r];
      S[row][col] = p;
      const int q = q0 + row;
      const int s = s_base + col;
      attn_out[((size_t)bh * Tn + q) * Tn + s] = p;
    }
  }
  __syncthreads();  // PV fragment gather reads other lanes' LDS writes

  // ---- pass 4: O = P @ V via WMMA (A = P from LDS, B = V^T contiguous) ----
  v8f oacc[4];
  for (int i = 0; i < 4; ++i)
    for (int j = 0; j < 8; ++j) oacc[i][j] = 0.0f;

  const int nblk = (ntiles + 1) >> 1;  // 32-key blocks (padding is zero)
  for (int blk = 0; blk < nblk; ++blk) {
    AFrag pf;
    const int prow = l15;
    const int cb = blk * 32 + hf * 16;
#pragma unroll
    for (int j = 0; j < 8; ++j) {
      pf.v[j]     = (bf16_t)S[prow][cb + j];
      pf.v[8 + j] = (bf16_t)S[prow][cb + 8 + j];
    }
#pragma unroll
    for (int nt = 0; nt < 4; ++nt) {
      const int d  = nt * 16 + l15;
      const int sr = s_base + blk * 32 + hf * 16;
      v16bf vv;
      if (sr < Tn) {
        vv = *(const v16bf*)(Vt + ((size_t)bh * DKn + d) * Tn + sr);
      } else {
        for (int j = 0; j < 16; ++j) vv[j] = (bf16_t)0.0f;
      }
      oacc[nt] = wmma_bf16(pf.v, vv, oacc[nt]);
    }
  }

  // write O as bf16 in (B*T, D) with column = h*64 + d (matches reshape)
#pragma unroll
  for (int nt = 0; nt < 4; ++nt) {
#pragma unroll
    for (int r = 0; r < 8; ++r) {
      const int q = q0 + r + 8 * hf;
      const int o = h * DKn + nt * 16 + l15;
      Obuf[((size_t)b * Tn + q) * Dn + o] = (bf16_t)oacc[nt][r];
    }
  }
}

// ------------------------------- launcher ----------------------------------
extern "C" void kernel_launch(void* const* d_in, const int* in_sizes, int n_in,
                              void* d_out, int out_size, void* d_ws, size_t ws_size,
                              hipStream_t stream) {
  const float* x  = (const float*)d_in[0];
  const float* Wq = (const float*)d_in[1];
  const float* Wk = (const float*)d_in[2];
  const float* Wv = (const float*)d_in[3];
  const float* Wo = (const float*)d_in[4];
  const int*   win = (const int*)d_in[5];

  float* out0 = (float*)d_out;                              // (B,T,D)
  float* attn = out0 + (size_t)Bn * Tn * Dn;                // (B,H,T,T)

  const size_t NX = (size_t)Bn * Tn * Dn;  // 4,194,304
  const size_t NW = (size_t)Dn * Dn;       // 1,048,576

  bf16_t* p = (bf16_t*)d_ws;
  bf16_t* xb  = p;  p += NX;
  bf16_t* Wqb = p;  p += NW;
  bf16_t* Wkb = p;  p += NW;
  bf16_t* Wvb = p;  p += NW;
  bf16_t* Wob = p;  p += NW;
  bf16_t* Qb  = p;  p += NX;
  bf16_t* Kb  = p;  p += NX;
  bf16_t* Vt  = p;  p += NX;
  bf16_t* Ob  = p;  p += NX;
  (void)ws_size; (void)n_in; (void)in_sizes; (void)out_size;

  // attention output is banded -> zero the full (B,H,T,T) region once per call
  hipMemsetAsync(attn, 0, (size_t)Bn * Hn * Tn * Tn * sizeof(float), stream);

  // f32 -> bf16 conversions
  cvt_f32_bf16<<<2048, 256, 0, stream>>>(x,  xb,  (int)NX);
  cvt_f32_bf16<<<1024, 256, 0, stream>>>(Wq, Wqb, (int)NW);
  cvt_f32_bf16<<<1024, 256, 0, stream>>>(Wk, Wkb, (int)NW);
  cvt_f32_bf16<<<1024, 256, 0, stream>>>(Wv, Wvb, (int)NW);
  cvt_f32_bf16<<<1024, 256, 0, stream>>>(Wo, Wob, (int)NW);

  // projections: (B*T)xD @ D x D^T
  const int M = Bn * Tn, N = Dn, K = Dn;
  dim3 gblk(32, 4);
  dim3 ggrd(M / 16, N / 256);
  proj_gemm<<<ggrd, gblk, 0, stream>>>(xb, Wqb, Qb, nullptr, M, N, K, 0);
  proj_gemm<<<ggrd, gblk, 0, stream>>>(xb, Wkb, Kb, nullptr, M, N, K, 0);
  proj_gemm<<<ggrd, gblk, 0, stream>>>(xb, Wvb, Vt, nullptr, M, N, K, 1);

  // banded ALiBi attention
  dim3 ablk(32, 2);
  dim3 agrd(Tn / 32, Hn, Bn);
  attn_kernel<<<agrd, ablk, 0, stream>>>(Qb, Kb, Vt, Ob, attn, win);

  // final projection straight to f32 output
  proj_gemm<<<ggrd, gblk, 0, stream>>>(Ob, Wob, nullptr, out0, M, N, K, 2);
}
